// DeTree_33612414059148
// MI455X (gfx1250) — matrix operand order repaired and checked by reading, not compiled, not run on hardware
//
#include <hip/hip_runtime.h>
#include <hip/hip_bf16.h>

typedef __bf16 bf16_t;
typedef __attribute__((ext_vector_type(16))) bf16_t v16bf;
typedef __attribute__((ext_vector_type(8)))  bf16_t v8bf;
typedef __attribute__((ext_vector_type(8)))  float  v8f;

constexpr int F = 512;          // IN_FEATURES
constexpr int T = 512;          // NUM_TREES
constexpr int D = 6;            // DEPTH
constexpr int R = 3;            // RESPONSE_DIM
constexpr int L = 64;           // NLEAF
constexpr int B = 1024;         // BATCH
constexpr int N_COLS   = T * D;     // 3072
constexpr int CW_PLANE = T * F;     // per-depth [t][f] plane (elements)
constexpr int RESP_PAD = 196;       // 192 + 4 pad: avoids 16-way LDS bank conflict

// ------------------------------------------------------------------
// Kernel A: x fp32 -> bf16 (row-major [B][F])
// ------------------------------------------------------------------
__global__ void cvt_x_bf16(const float* __restrict__ x,
                           bf16_t* __restrict__ xb, int n) {
  int i = blockIdx.x * blockDim.x + threadIdx.x;
  if (i < n) xb[i] = (bf16_t)x[i];
}

// ------------------------------------------------------------------
// Kernel B: softmax over features (axis 0) of feat_attention[F][T*D],
// written transposed & depth-planar: cwT[d][t][f] (bf16, f contiguous)
// One thread per column; reads are coalesced across threads per row.
// ------------------------------------------------------------------
__global__ void softmax_cols(const float* __restrict__ fa,
                             bf16_t* __restrict__ cwT) {
  int c = blockIdx.x * blockDim.x + threadIdx.x;  // 0..3071
  if (c >= N_COLS) return;
  // pass 1: online max & sum
  float m = -3.402823466e38f, s = 0.0f;
  for (int f = 0; f < F; ++f) {
    float v  = fa[f * N_COLS + c];
    float nm = fmaxf(m, v);
    s = s * __expf(m - nm) + __expf(v - nm);
    m = nm;
  }
  float inv = 1.0f / s;
  int t = c / D, d = c % D;
  bf16_t* dst = cwT + (size_t)d * CW_PLANE + (size_t)t * F;
  // pass 2: normalized exp -> bf16
  for (int f = 0; f < F; ++f) {
    float v = fa[f * N_COLS + c];
    dst[f] = (bf16_t)(__expf(v - m) * inv);
  }
}

// ------------------------------------------------------------------
// helpers for fragment loads
// ------------------------------------------------------------------
__device__ __forceinline__ v16bf load_a_frag(const bf16_t* p) {
  v8bf lo = *(const v8bf*)(p);
  v8bf hi = *(const v8bf*)(p + 8);
  return __builtin_shufflevector(lo, hi,
      0, 1, 2, 3, 4, 5, 6, 7, 8, 9, 10, 11, 12, 13, 14, 15);
}

// ------------------------------------------------------------------
// Kernel C: fused GEMM (bf16 WMMA, f32 acc) + sparsemoid bins +
// oblivious-tree leaf products + response contraction.
// Block = 256 threads = 8 waves. blockIdx.x -> tree tile (16 trees),
// blockIdx.y*8 + wave -> batch tile (16 rows). 6 accumulators/wave
// (one per depth); 96 v_wmma per wave (dynamic).
// Manually software-pipelined (double-buffered A + 6x B fragments):
// next K-slab's 14 loads are issued before the current slab's 6
// WMMAs, so each WMMA waits on loads a full iteration old (~12 deep)
// instead of the s_wait_loadcnt 0x0 the scheduler produced on its own.
// ~180 VGPRs: below the 256 MSB boundary; grid (2048 waves) limits
// occupancy to ~4 waves/SIMD anyway.
// ------------------------------------------------------------------
__launch_bounds__(256)
__global__ void detree_fused(const bf16_t* __restrict__ xb,
                             const bf16_t* __restrict__ cwT,
                             const float*  __restrict__ thresholds,
                             const float*  __restrict__ logT,
                             const float*  __restrict__ response,
                             float* __restrict__ out) {
  __shared__ float respLds[16 * RESP_PAD];   // 16 trees x 64 leaves x 3, padded

  const int treeBase = blockIdx.x * 16;
  const int wave = threadIdx.x >> 5;
  const int lane = threadIdx.x & 31;
  const int mBase = (blockIdx.y * 8 + wave) * 16;

  // cooperative stage of this block's response tile into LDS
  const float* rsrc = response + (size_t)treeBase * (L * R);
  for (int i = threadIdx.x; i < 16 * L * R; i += 256) {
    int tr = i / (L * R), w = i - tr * (L * R);
    respLds[tr * RESP_PAD + w] = rsrc[i];
  }
  __syncthreads();

  const int n   = lane & 15;   // tree column in tile / A row in tile
  const int sel = lane >> 4;   // K half selector
  const int t   = treeBase + n;

  v8f acc[6];
#pragma unroll
  for (int d = 0; d < 6; ++d) { v8f z = {}; acc[d] = z; }

  // A: lane (M = n), K contiguous 16 per half
  const bf16_t* aBase = xb + (size_t)(mBase + n) * F + sel * 16;
  // B: lane (N = n), K contiguous 16 per half
  const bf16_t* bBase = cwT + (size_t)t * F + sel * 16;

  // ---- software-pipelined K loop (double-buffered fragments) ----
  v16bf aN = load_a_frag(aBase);
  v16bf bN[6];
#pragma unroll
  for (int d = 0; d < 6; ++d)
    bN[d] = *(const v16bf*)(bBase + (size_t)d * CW_PLANE);

#pragma unroll 3
  for (int kk = 0; kk < F - 32; kk += 32) {
    v16bf aC = aN;
    v16bf bC[6];
#pragma unroll
    for (int d = 0; d < 6; ++d) bC[d] = bN[d];

    // issue ALL next-slab loads before this slab's WMMAs
    aN = load_a_frag(aBase + kk + 32);
#pragma unroll
    for (int d = 0; d < 6; ++d)
      bN[d] = *(const v16bf*)(bBase + (size_t)d * CW_PLANE + kk + 32);
    if (kk + 64 < F) {
      __builtin_prefetch(aBase + kk + 64, 0, 1);          // global_prefetch_b8
      __builtin_prefetch(bBase + kk + 64, 0, 1);
    }

#pragma unroll
    for (int d = 0; d < 6; ++d)
      acc[d] = __builtin_amdgcn_wmma_f32_16x16x32_bf16(
          false, aC, false, bC[d], (short)0, acc[d], false, false);
  }
  // drain: last slab
#pragma unroll
  for (int d = 0; d < 6; ++d)
    acc[d] = __builtin_amdgcn_wmma_f32_16x16x32_bf16(
        false, aN, false, bN[d], (short)0, acc[d], false, false);

  // ---- epilogue: bins -> leaf weights -> response contraction ----
  float thr[6], sc[6];
#pragma unroll
  for (int d = 0; d < 6; ++d) {
    thr[d] = thresholds[t * 6 + d];
    sc[d]  = __expf(-logT[t * 6 + d]);
  }
  const float* rl = &respLds[n * RESP_PAD];

#pragma unroll
  for (int i = 0; i < 8; ++i) {
    // C layout: VGPR i -> row i (lanes 0-15) / row i+8 (lanes 16-31)
    const int brow = mBase + i + sel * 8;
    float g0[6], g1[6];
#pragma unroll
    for (int d = 0; d < 6; ++d) {
      float fv = acc[d][i];
      float bn = 0.5f * (fv - thr[d]) * sc[d] + 0.5f;   // sparsemoid
      bn = fminf(fmaxf(bn, 0.0f), 1.0f);
      g0[d] = bn; g1[d] = 1.0f - bn;                    // gate(2j) / gate(2j+1)
    }
    // factorized leaf products: leaf = lo(bits 0..2) + 8*hi(bits 3..5)
    float plo[8];
#pragma unroll
    for (int lo = 0; lo < 8; ++lo)
      plo[lo] = ((lo & 1) ? g1[0] : g0[0]) *
                ((lo & 2) ? g1[1] : g0[1]) *
                ((lo & 4) ? g1[2] : g0[2]);
    float o0 = 0.f, o1 = 0.f, o2 = 0.f;
#pragma unroll
    for (int hb = 0; hb < 8; ++hb) {
      float whi = ((hb & 1) ? g1[3] : g0[3]) *
                  ((hb & 2) ? g1[4] : g0[4]) *
                  ((hb & 4) ? g1[5] : g0[5]);
#pragma unroll
      for (int lo = 0; lo < 8; ++lo) {
        float w = whi * plo[lo];
        const float* rp = rl + (hb * 8 + lo) * 3;
        o0 += w * rp[0]; o1 += w * rp[1]; o2 += w * rp[2];
      }
    }
    float* op = out + (size_t)brow * (T * R) + t * R;
    op[0] = o0; op[1] = o1; op[2] = o2;
  }
}

// ------------------------------------------------------------------
extern "C" void kernel_launch(void* const* d_in, const int* in_sizes, int n_in,
                              void* d_out, int out_size, void* d_ws, size_t ws_size,
                              hipStream_t stream) {
  const float* x    = (const float*)d_in[0];  // [1024,512]
  const float* fa   = (const float*)d_in[1];  // [512,3072]
  const float* thr  = (const float*)d_in[2];  // [512,6]
  const float* lt   = (const float*)d_in[3];  // [512,6]
  const float* resp = (const float*)d_in[4];  // [512,64,3]
  // d_in[5] = path_map: structure is static (oblivious tree), hardcoded.
  float* out = (float*)d_out;                 // [1024, 1536]

  bf16_t* xb  = (bf16_t*)d_ws;                                   // 1 MB
  bf16_t* cwT = (bf16_t*)((char*)d_ws + (size_t)B * F * 2);      // 3 MB

  cvt_x_bf16<<<(B * F + 255) / 256, 256, 0, stream>>>(x, xb, B * F);
  softmax_cols<<<(N_COLS + 255) / 256, 256, 0, stream>>>(fa, cwT);

  dim3 grid(T / 16, B / (16 * 8));   // (32, 8), 256 threads each
  detree_fused<<<grid, 256, 0, stream>>>(xb, cwT, thr, lt, resp, out);
}
